// Net_6519760355363
// MI455X (gfx1250) — compile-verified, matrix operand order checked by reading, not run on hardware
//
#include <hip/hip_runtime.h>
#include <hip/hip_bf16.h>

typedef __attribute__((ext_vector_type(16))) _Float16 v16h;
typedef __attribute__((ext_vector_type(8)))  float    v8f;

#define CDIV(a,b) (((a)+(b)-1)/(b))

// ---------------------------------------------------------------------------
// WMMA fragment load: 16x32 f16 tile, row-major source with leading dim `ld`.
// Per CDNA5 ISA 16-bit A-matrix layout: lane L -> row L%16; per lane the 16
// halves live in two contiguous 16B chunks at byte offsets g*16 and 32+g*16
// (g = L/16), so the compiler merges this into 2x global_load_b128.
// B operands are stored TRANSPOSED [N x Kp] so the same loader serves both.
// ---------------------------------------------------------------------------
__device__ __forceinline__ v16h load_frag16x32(const _Float16* __restrict__ p, int ld) {
  const int lane = threadIdx.x & 31;
  const int r = lane & 15, g = lane >> 4;
  const _Float16* q = p + (long long)r * ld;
  v16h f;
#pragma unroll
  for (int j = 0; j < 8; ++j) {
    const int kk = ((j < 4) ? 0 : 16) + (g << 3) + ((j & 3) << 1);
    f[2 * j]     = q[kk];
    f[2 * j + 1] = q[kk + 1];
  }
  return f;
}

// ---------------------------------------------------------------------------
// Strided-batched GEMM: C[M x N] = A[M x Kp] * Bt[Np x Kp]^T
// 4 waves / block; each wave computes 16(M) x (16*JT)(N) with JT accumulators.
// JT is a compile-time constant -> branch-free straight-line K-loop, so the
// scheduler can overlap next-iteration b128 loads with current WMMAs.
// Kp multiple of 32; column padding handled by zero-padded Bt rows.
// Optional epilogue: per-column bias + ReLU.
// ---------------------------------------------------------------------------
template<int JT>
__global__ __launch_bounds__(128) void wmma_gemm_t(
    const _Float16* __restrict__ A, const _Float16* __restrict__ Bt,
    float* __restrict__ C, int M, int N, int Kp, int n_base,
    long long sA, long long sB, long long sC,
    const float* __restrict__ bias, int relu)
{
  const int bz = blockIdx.z;
  A  += (long long)bz * sA;
  Bt += (long long)bz * sB;
  C  += (long long)bz * sC;
  const int wave = threadIdx.x >> 5;
  const int lane = threadIdx.x & 31;
  const int m0 = (blockIdx.y * 4 + wave) << 4;
  const int n0 = n_base + (blockIdx.x << 6);
  if (m0 >= M) return;                      // wave-uniform: EXEC stays all-ones

  v8f acc[JT];
#pragma unroll
  for (int j = 0; j < JT; ++j)
#pragma unroll
    for (int r = 0; r < 8; ++r) acc[j][r] = 0.f;

  const _Float16* Ap = A  + (long long)m0 * Kp;
  const _Float16* Bp = Bt + (long long)n0 * Kp;
  const int kit = Kp >> 5;
#pragma unroll 2
  for (int t = 0; t < kit; ++t) {
    const int k = t << 5;
    v16h a = load_frag16x32(Ap + k, Kp);
#pragma unroll
    for (int j = 0; j < JT; ++j) {
      v16h bf = load_frag16x32(Bp + (long long)(j << 4) * Kp + k, Kp);
      acc[j] = __builtin_amdgcn_wmma_f32_16x16x32_f16(
          false, a, false, bf, (short)0, acc[j], false, false);
    }
  }

  // C/D layout: VGPR r -> M = m0 + (lane/16)*8 + r ; N = n0 + 16j + lane%16
  const int cn = lane & 15, cg = lane >> 4;
#pragma unroll
  for (int j = 0; j < JT; ++j) {
    const int nn = n0 + (j << 4) + cn;
    if (nn < N) {
      const float bb = bias ? bias[nn] : 0.f;
#pragma unroll
      for (int r = 0; r < 8; ++r) {
        const int mm = m0 + (cg << 3) + r;
        float v = acc[j][r] + bb;
        if (relu) v = fmaxf(v, 0.f);
        C[(long long)mm * N + nn] = v;
      }
    }
  }
}

// Host-side launcher: full 64-wide column blocks with JT=4, plus one
// specialized remainder launch for the trailing (Np%64)/16 tiles.
static void launch_gemm(hipStream_t stream,
                        const _Float16* A, const _Float16* Bt, float* C,
                        int M, int N, int Np, int Kp,
                        long long sA, long long sB, long long sC,
                        const float* bias, int relu, int batch) {
  const int gy   = CDIV(M, 64);
  const int full = Np >> 6;
  const int rem  = (Np & 63) >> 4;
  if (full > 0) {
    dim3 g(full, gy, batch);
    wmma_gemm_t<4><<<g, 128, 0, stream>>>(A, Bt, C, M, N, Kp, 0, sA, sB, sC, bias, relu);
  }
  if (rem) {
    dim3 g(1, gy, batch);
    const int nb = full << 6;
    switch (rem) {
      case 1: wmma_gemm_t<1><<<g, 128, 0, stream>>>(A, Bt, C, M, N, Kp, nb, sA, sB, sC, bias, relu); break;
      case 2: wmma_gemm_t<2><<<g, 128, 0, stream>>>(A, Bt, C, M, N, Kp, nb, sA, sB, sC, bias, relu); break;
      default:wmma_gemm_t<3><<<g, 128, 0, stream>>>(A, Bt, C, M, N, Kp, nb, sA, sB, sC, bias, relu); break;
    }
  }
}

// ---------------------------------------------------------------------------
// Packing / conversion kernels (f32 -> zero-padded f16 operands)
// ---------------------------------------------------------------------------
__global__ void pack_a(const float* __restrict__ src, int M, int K, int Kp,
                       _Float16* __restrict__ dst, int total) {
  int idx = blockIdx.x * blockDim.x + threadIdx.x;
  if (idx >= total) return;
  int m = idx / Kp, k = idx - m * Kp;
  dst[idx] = (k < K) ? (_Float16)src[(long long)m * K + k] : (_Float16)0.f;
  (void)M;
}

__global__ void pack_b(const float* __restrict__ src, int K, int N, int Kp, int Np,
                       _Float16* __restrict__ dst, int total) {
  int idx = blockIdx.x * blockDim.x + threadIdx.x;
  if (idx >= total) return;
  int n = idx / Kp, k = idx - n * Kp;
  dst[idx] = (n < N && k < K) ? (_Float16)src[(long long)k * N + n] : (_Float16)0.f;
  (void)Np;
}

// cw1 (128,2000,3) -> w1t [i][o*3+k]  (row i contiguous for the scatter kernel)
__global__ void cw1_transpose(const float* __restrict__ cw1, float* __restrict__ w1t, int total) {
  int idx = blockIdx.x * blockDim.x + threadIdx.x;
  if (idx >= total) return;
  int i = idx / 384, ok = idx - i * 384;
  int o = ok / 3, k = ok - o * 3;
  w1t[idx] = cw1[(long long)o * 6000 + i * 3 + k];
}

// EmbShift Bt [l(128 rows, pad from 126) x j(96 cols, pad from 66)]; j = v*3+k
__global__ void build_embT(const float* __restrict__ emb, _Float16* __restrict__ out) {
  int idx = blockIdx.x * blockDim.x + threadIdx.x;
  if (idx >= 128 * 96) return;
  int l = idx / 96, j = idx - l * 96;
  float v = 0.f;
  if (l < 126 && j < 66) v = emb[(j / 3) * 128 + l + (j % 3)];
  out[idx] = (_Float16)v;
}

// ---------------------------------------------------------------------------
// conv1 bucket trick: S[b,v,ok] = sum_{i: target[b,i]==v} w1t[i][ok]
// Thread t owns LDS column t -> race-free; 22*384 f32 + 2000 int in LDS (~42KB,
// comfortably inside the 320KB WGP LDS). Emits A-operand [128(o) x 96(j)] f16.
// ---------------------------------------------------------------------------
__global__ __launch_bounds__(384) void conv1_scatter(
    const int* __restrict__ target, const float* __restrict__ w1t,
    _Float16* __restrict__ S16)
{
  __shared__ float S[22 * 384];
  __shared__ int   tg[2000];
  const int b = blockIdx.x;
  const int t = threadIdx.x;
  for (int i = t; i < 22 * 384; i += 384) S[i] = 0.f;
  for (int i = t; i < 2000; i += 384) tg[i] = target[b * 2000 + i];
  __syncthreads();
  for (int i = 0; i < 2000; ++i) {
    const int v = tg[i];
    S[v * 384 + t] += w1t[(long long)i * 384 + t];
  }
  __syncthreads();
  _Float16* out = S16 + (long long)b * 128 * 96;
  for (int idx = t; idx < 128 * 96; idx += 384) {
    const int o = idx / 96, j = idx - o * 96;
    float val = 0.f;
    if (j < 66) val = S[(j / 3) * 384 + o * 3 + (j % 3)];
    out[idx] = (_Float16)val;
  }
}

// transposed im2col: out[b][l(pad Lp)][i*KK+k] = P[b,i,l+k]  (f16 Bt operand)
__global__ void im2col_t(const float* __restrict__ P, int Bn, int Cin, int Lin,
                         int KK, int Lout, int Lp, _Float16* __restrict__ out, int total) {
  int idx = blockIdx.x * blockDim.x + threadIdx.x;
  if (idx >= total) return;
  int per_b = Lp * Cin * KK;
  int b = idx / per_b;
  int rem = idx - b * per_b;
  int l = rem / (Cin * KK);
  int j = rem - l * (Cin * KK);
  int i = j / KK, k = j - i * KK;
  float v = 0.f;
  if (l < Lout) v = P[((long long)b * Cin + i) * Lin + l + k];
  out[idx] = (_Float16)v;
  (void)Bn;
}

// max_pool1d(2,2) + bias(per out-channel) + relu
__global__ void pool_bias_relu(const float* __restrict__ Cin_, const float* __restrict__ bias,
                               int Bn, int Och, int Lin, int Lout,
                               float* __restrict__ P, int total) {
  int idx = blockIdx.x * blockDim.x + threadIdx.x;
  if (idx >= total) return;
  int l = idx % Lout;
  int bo = idx / Lout;
  int o = bo % Och;
  const float* row = Cin_ + (long long)bo * Lin;
  float v = fmaxf(row[2 * l], row[2 * l + 1]) + bias[o];
  P[idx] = fmaxf(v, 0.f);
  (void)Bn;
}

// ---------------------------------------------------------------------------
// GAT pieces
// ---------------------------------------------------------------------------
__device__ __forceinline__ unsigned fenc(float f) {
  unsigned u = __float_as_uint(f);
  return (u & 0x80000000u) ? ~u : (u | 0x80000000u);
}
__device__ __forceinline__ float fdec(unsigned u) {
  return __uint_as_float((u & 0x80000000u) ? (u & 0x7FFFFFFFu) : ~u);
}

__global__ void gat_dots(const float* __restrict__ h, const float* __restrict__ as_,
                         const float* __restrict__ ad_, int Nn, int H, int C,
                         float* __restrict__ s, float* __restrict__ d, int total) {
  int idx = blockIdx.x * blockDim.x + threadIdx.x;
  if (idx >= total) return;
  int n = idx / H, hh = idx - n * H;
  const float* hp = h + ((long long)n * H + hh) * C;
  float ss = 0.f, dd = 0.f;
  for (int c = 0; c < C; ++c) { ss += hp[c] * as_[hh * C + c]; dd += hp[c] * ad_[hh * C + c]; }
  s[idx] = ss; d[idx] = dd;
  (void)Nn;
}

__global__ void edge_max_k(const int* __restrict__ ei, int E, int Nn, int H,
                           const float* __restrict__ s, const float* __restrict__ d,
                           unsigned* __restrict__ m, int total) {
  int idx = blockIdx.x * blockDim.x + threadIdx.x;
  if (idx >= total) return;
  int e = idx / H, hh = idx - e * H;
  int src = (e < E) ? ei[e] : (e - E);
  int dst = (e < E) ? ei[E + e] : (e - E);
  float v = s[src * H + hh] + d[dst * H + hh];
  v = (v > 0.f) ? v : 0.2f * v;                         // leaky_relu(0.2)
  atomicMax(&m[dst * H + hh], fenc(v));
  (void)Nn;
}

__global__ void edge_expsum(const int* __restrict__ ei, int E, int Nn, int H,
                            const float* __restrict__ s, const float* __restrict__ d,
                            const unsigned* __restrict__ m,
                            float* __restrict__ exbuf, float* __restrict__ denom, int total) {
  int idx = blockIdx.x * blockDim.x + threadIdx.x;
  if (idx >= total) return;
  int e = idx / H, hh = idx - e * H;
  int src = (e < E) ? ei[e] : (e - E);
  int dst = (e < E) ? ei[E + e] : (e - E);
  float v = s[src * H + hh] + d[dst * H + hh];
  v = (v > 0.f) ? v : 0.2f * v;
  float ex = expf(v - fdec(m[dst * H + hh]));
  exbuf[idx] = ex;
  atomicAdd(&denom[dst * H + hh], ex);
  (void)Nn;
}

__global__ void edge_aggregate(const int* __restrict__ ei, int E, int Nn, int H, int C,
                               const float* __restrict__ exbuf, const float* __restrict__ denom,
                               const float* __restrict__ h, float* __restrict__ outacc, int total) {
  int idx = blockIdx.x * blockDim.x + threadIdx.x;
  if (idx >= total) return;
  int D = H * C;
  int e = idx / D, f = idx - e * D;
  int hh = f / C;
  int src = (e < E) ? ei[e] : (e - E);
  int dst = (e < E) ? ei[E + e] : (e - E);
  float alpha = exbuf[e * H + hh] / denom[dst * H + hh];
  atomicAdd(&outacc[(long long)dst * D + f], h[(long long)src * D + f] * alpha);
  (void)Nn;
}

// bias + relu; always emits f16 A-operand [Nr x Kp], optionally mirrors f32
__global__ void bias_relu_pack(const float* __restrict__ acc, const float* __restrict__ bias,
                               int Nr, int D, int Kp, _Float16* __restrict__ out16,
                               float* __restrict__ out32, int total) {
  int idx = blockIdx.x * blockDim.x + threadIdx.x;
  if (idx >= total) return;
  int r = idx / Kp, c = idx - r * Kp;
  float v = 0.f;
  if (c < D) {
    v = fmaxf(acc[(long long)r * D + c] + bias[c], 0.f);
    if (out32) out32[(long long)r * D + c] = v;
  }
  out16[idx] = (_Float16)v;
  (void)Nr;
}

// global max pool over sorted batch (npg nodes/graph) -> f16 padded A operand
__global__ void graph_pool_pack(const float* __restrict__ h, int Bn, int npg, int D, int Kp,
                                _Float16* __restrict__ out, int total) {
  int idx = blockIdx.x * blockDim.x + threadIdx.x;
  if (idx >= total) return;
  int b = idx / Kp, c = idx - b * Kp;
  float v = 0.f;
  if (c < D) {
    v = -3.0e38f;
    for (int n = 0; n < npg; ++n)
      v = fmaxf(v, h[((long long)b * npg + n) * D + c]);
  }
  out[idx] = (_Float16)v;
  (void)Bn;
}

__global__ void concat_pack(const float* __restrict__ g2, const float* __restrict__ t2,
                            _Float16* __restrict__ out, int total) {
  int idx = blockIdx.x * blockDim.x + threadIdx.x;
  if (idx >= total) return;
  int b = idx >> 7, f = idx & 127;
  float v = (f < 64) ? g2[b * 64 + f] : t2[b * 64 + (f - 64)];
  out[idx] = (_Float16)v;
}

__global__ void final_out(const float* __restrict__ D2, const float* __restrict__ ow,
                          const float* __restrict__ ob, float* __restrict__ out, int Bn) {
  int b = blockIdx.x * blockDim.x + threadIdx.x;
  if (b >= Bn) return;
  float a = ob[0];
  for (int f = 0; f < 64; ++f) a += D2[b * 64 + f] * ow[f];
  out[b] = a;
}

// ---------------------------------------------------------------------------
// Host driver
// ---------------------------------------------------------------------------
extern "C" void kernel_launch(void* const* d_in, const int* in_sizes, int n_in,
                              void* d_out, int out_size, void* d_ws, size_t ws_size,
                              hipStream_t stream) {
  const float* x     = (const float*)d_in[0];
  const int*   ei    = (const int*)d_in[1];
  const int*   target= (const int*)d_in[3];
  const float* W1 = (const float*)d_in[4],  *as1 = (const float*)d_in[5];
  const float* ad1= (const float*)d_in[6],  *b1  = (const float*)d_in[7];
  const float* W2 = (const float*)d_in[8],  *as2 = (const float*)d_in[9];
  const float* ad2= (const float*)d_in[10], *b2  = (const float*)d_in[11];
  const float* W3 = (const float*)d_in[12], *as3 = (const float*)d_in[13];
  const float* ad3= (const float*)d_in[14], *b3  = (const float*)d_in[15];
  const float* fg1w=(const float*)d_in[16], *fg1b=(const float*)d_in[17];
  const float* fg2w=(const float*)d_in[18], *fg2b=(const float*)d_in[19];
  const float* emb =(const float*)d_in[20];
  const float* cw1 =(const float*)d_in[21], *cb1 =(const float*)d_in[22];
  const float* cw2 =(const float*)d_in[23], *cb2 =(const float*)d_in[24];
  const float* cw3 =(const float*)d_in[25], *cb3 =(const float*)d_in[26];
  const float* fx1w=(const float*)d_in[27], *fx1b=(const float*)d_in[28];
  const float* fx2w=(const float*)d_in[29], *fx2b=(const float*)d_in[30];
  const float* f12w=(const float*)d_in[31], *f12b=(const float*)d_in[32];
  const float* f22w=(const float*)d_in[33], *f22b=(const float*)d_in[34];
  const float* ow  =(const float*)d_in[35], *ob  =(const float*)d_in[36];

  const int Nn   = in_sizes[0] / 78;     // 51200
  const int E    = in_sizes[1] / 2;      // 204800
  const int Bb   = in_sizes[3] / 2000;   // 1024
  const int Etot = E + Nn;               // self-loops appended
  const int npg  = Nn / Bb;              // 50 nodes/graph (sorted batch)
  (void)n_in; (void)out_size; (void)ws_size;

  // ---- bump allocator over d_ws with phase-based aliasing of big buffers ----
  char* base = (char*)d_ws; size_t off = 0;
  auto alloc = [&](size_t bytes) -> void* {
    off = (off + 255) & ~(size_t)255;
    void* p = base + off; off += bytes; return p;
  };
  size_t szR1 = (size_t)Bb * 128 * 126 * 4;                 // C1
  if ((size_t)Nn * 312 * 4 > szR1) szR1 = (size_t)Nn * 312 * 4;
  float* bufG = (float*)alloc(szR1);                        // GAT gemm out / C1
  size_t szR2 = (size_t)Bb * 32 * 1024 * 2;                 // Im3t
  if ((size_t)Nn * 312 * 4 > szR2) szR2 = (size_t)Nn * 312 * 4;
  float* bufO = (float*)alloc(szR2);                        // GAT agg accum / Im3t
  _Float16* Im3t = (_Float16*)bufO;
  size_t szR3 = (size_t)Bb * 64 * 640 * 2;                  // Im2t
  if ((size_t)Nn * 320 * 2 > szR3) szR3 = (size_t)Nn * 320 * 2;
  _Float16* h16 = (_Float16*)alloc(szR3);                   // packed activations / Im2t
  _Float16* Im2t = h16;
  size_t szR4 = (size_t)Bb * 128 * 63 * 4;                  // P1 (>= C2)
  float* P1 = (float*)alloc(szR4);  float* C2 = P1;
  size_t szR5 = (size_t)Bb * 128 * 96 * 2;                  // S16 (>= x16)
  if ((size_t)Nn * 96 * 2 > szR5) szR5 = (size_t)Nn * 96 * 2;
  _Float16* x16 = (_Float16*)alloc(szR5);  _Float16* S16 = x16;

  float*    sdot = (float*)alloc((size_t)Nn * 2 * 4);
  float*    ddot = (float*)alloc((size_t)Nn * 2 * 4);
  unsigned* mbuf = (unsigned*)alloc((size_t)Nn * 2 * 4);
  float*    denom= (float*)alloc((size_t)Nn * 2 * 4);
  float*    exbuf= (float*)alloc((size_t)Etot * 2 * 4);
  _Float16* W1t  = (_Float16*)alloc(160 * 96 * 2);
  _Float16* W2t  = (_Float16*)alloc(240 * 160 * 2);
  _Float16* W3t  = (_Float16*)alloc(320 * 256 * 2);
  _Float16* fg1t = (_Float16*)alloc(256 * 320 * 2);
  _Float16* fg2t = (_Float16*)alloc(64 * 256 * 2);
  _Float16* fx1t = (_Float16*)alloc(256 * 352 * 2);
  _Float16* fx2t = (_Float16*)alloc(64 * 256 * 2);
  _Float16* f12t = (_Float16*)alloc(128 * 128 * 2);
  _Float16* f22t = (_Float16*)alloc(64 * 128 * 2);
  _Float16* W2c  = (_Float16*)alloc(128 * 640 * 2);
  _Float16* W3c  = (_Float16*)alloc(32 * 1024 * 2);
  _Float16* embT = (_Float16*)alloc(128 * 96 * 2);
  float*    w1t  = (float*)alloc((size_t)2000 * 384 * 4);
  _Float16* g16  = (_Float16*)alloc((size_t)Bb * 320 * 2);
  float*    F1   = (float*)alloc((size_t)Bb * 256 * 4);
  _Float16* F1h  = (_Float16*)alloc((size_t)Bb * 256 * 2);
  float*    gvec = (float*)alloc((size_t)Bb * 64 * 4);
  float*    P2   = (float*)alloc((size_t)Bb * 128 * 29 * 4);
  float*    C3   = (float*)alloc((size_t)Bb * 32 * 22 * 4);
  float*    P3   = (float*)alloc((size_t)Bb * 352 * 4);
  _Float16* t352h= (_Float16*)alloc((size_t)Bb * 352 * 2);
  float*    TX1  = (float*)alloc((size_t)Bb * 256 * 4);
  _Float16* TX1h = (_Float16*)alloc((size_t)Bb * 256 * 2);
  float*    tvec = (float*)alloc((size_t)Bb * 64 * 4);
  _Float16* c16  = (_Float16*)alloc((size_t)Bb * 128 * 2);
  float*    D1   = (float*)alloc((size_t)Bb * 128 * 4);
  _Float16* D1h  = (_Float16*)alloc((size_t)Bb * 128 * 2);
  float*    D2   = (float*)alloc((size_t)Bb * 64 * 4);

  const int TB = 256;
#define EL(n) dim3((unsigned)CDIV((n), TB)), dim3(TB), 0, stream

  // ---- operand packing (deterministic every call) ----
  int n;
  n = Nn * 96;   pack_a<<<EL(n)>>>(x,    Nn, 78,   96,  x16, n);
  n = 160 * 96;  pack_b<<<EL(n)>>>(W1,   78, 156,  96,  160, W1t, n);
  n = 240 * 160; pack_b<<<EL(n)>>>(W2,  156, 234, 160,  240, W2t, n);
  n = 320 * 256; pack_b<<<EL(n)>>>(W3,  234, 312, 256,  320, W3t, n);
  n = 256 * 320; pack_b<<<EL(n)>>>(fg1w,312, 256, 320,  256, fg1t, n);
  n = 64 * 256;  pack_b<<<EL(n)>>>(fg2w,256,  64, 256,   64, fg2t, n);
  n = 256 * 352; pack_b<<<EL(n)>>>(fx1w,352, 256, 352,  256, fx1t, n);
  n = 64 * 256;  pack_b<<<EL(n)>>>(fx2w,256,  64, 256,   64, fx2t, n);
  n = 128 * 128; pack_b<<<EL(n)>>>(f12w,128, 128, 128,  128, f12t, n);
  n = 64 * 128;  pack_b<<<EL(n)>>>(f22w,128,  64, 128,   64, f22t, n);
  n = 128 * 640; pack_a<<<EL(n)>>>(cw2, 128, 640, 640,  W2c, n);
  n = 32 * 1024; pack_a<<<EL(n)>>>(cw3,  32,1024,1024,  W3c, n);
  n = 128 * 96;  build_embT<<<EL(n)>>>(emb, embT);
  n = 2000 * 384; cw1_transpose<<<EL(n)>>>(cw1, w1t, n);

  // ---- GAT layers ----
  auto gat_layer = [&](const _Float16* Ain, const _Float16* Wt, int Np, int Kp,
                       const float* a_s, const float* a_d, const float* bias,
                       int H, int Cc, int Kp_next, float* out32) {
    const int Dout = H * Cc;
    launch_gemm(stream, Ain, Wt, bufG, Nn, Dout, Np, Kp, 0, 0, 0, nullptr, 0, 1);
    int t1 = Nn * H;
    gat_dots<<<EL(t1)>>>(bufG, a_s, a_d, Nn, H, Cc, sdot, ddot, t1);
    hipMemsetAsync(mbuf,  0, (size_t)Nn * H * 4, stream);
    hipMemsetAsync(denom, 0, (size_t)Nn * H * 4, stream);
    hipMemsetAsync(bufO,  0, (size_t)Nn * Dout * 4, stream);
    int t2 = Etot * H;
    edge_max_k<<<EL(t2)>>>(ei, E, Nn, H, sdot, ddot, mbuf, t2);
    edge_expsum<<<EL(t2)>>>(ei, E, Nn, H, sdot, ddot, mbuf, exbuf, denom, t2);
    int t3 = Etot * Dout;
    edge_aggregate<<<EL(t3)>>>(ei, E, Nn, H, Cc, exbuf, denom, bufG, bufO, t3);
    int t4 = Nn * Kp_next;
    bias_relu_pack<<<EL(t4)>>>(bufO, bias, Nn, Dout, Kp_next, h16, out32, t4);
  };
  gat_layer(x16, W1t, 160,  96, as1, ad1, b1, 2,  78, 160, nullptr);
  gat_layer(h16, W2t, 240, 160, as2, ad2, b2, 1, 234, 256, nullptr);
  gat_layer(h16, W3t, 320, 256, as3, ad3, b3, 1, 312, 320, bufG);   // f32 h3 -> bufG

  // ---- graph branch: global max pool + 2 FC ----
  n = Bb * 320; graph_pool_pack<<<EL(n)>>>(bufG, Bb, npg, 312, 320, g16, n);
  launch_gemm(stream, g16, fg1t, F1, Bb, 256, 256, 320, 0, 0, 0, fg1b, 1, 1);
  n = Bb * 256; pack_a<<<EL(n)>>>(F1, Bb, 256, 256, F1h, n);
  launch_gemm(stream, F1h, fg2t, gvec, Bb, 64, 64, 256, 0, 0, 0, fg2b, 1, 1);

  // ---- target branch: conv1 via 22-bucket trick, conv2/conv3 implicit GEMM ----
  conv1_scatter<<<dim3(Bb), dim3(384), 0, stream>>>(target, w1t, S16);
  // C1[b] = S16[b](128x96) * embT(128x96)^T -> [128 x 126]
  launch_gemm(stream, S16, embT, bufG, 128, 126, 128, 96,
              (long long)128 * 96, 0, (long long)128 * 126, nullptr, 0, Bb);
  n = Bb * 128 * 63; pool_bias_relu<<<EL(n)>>>(bufG, cb1, Bb, 128, 126, 63, P1, n);
  n = Bb * 64 * 640; im2col_t<<<EL(n)>>>(P1, Bb, 128, 63, 5, 59, 64, Im2t, n);
  // C2[b] = W2c(128x640) * Im2t[b](64x640)^T -> [128 x 59]
  launch_gemm(stream, W2c, Im2t, C2, 128, 59, 64, 640,
              0, (long long)64 * 640, (long long)128 * 59, nullptr, 0, Bb);
  n = Bb * 128 * 29; pool_bias_relu<<<EL(n)>>>(C2, cb2, Bb, 128, 59, 29, P2, n);
  n = Bb * 32 * 1024; im2col_t<<<EL(n)>>>(P2, Bb, 128, 29, 8, 22, 32, Im3t, n);
  // C3[b] = W3c(32x1024) * Im3t[b](32x1024)^T -> [32 x 22]
  launch_gemm(stream, W3c, Im3t, C3, 32, 22, 32, 1024,
              0, (long long)32 * 1024, (long long)32 * 22, nullptr, 0, Bb);
  n = Bb * 32 * 11; pool_bias_relu<<<EL(n)>>>(C3, cb3, Bb, 32, 22, 11, P3, n);
  n = Bb * 352; pack_a<<<EL(n)>>>(P3, Bb, 352, 352, t352h, n);
  launch_gemm(stream, t352h, fx1t, TX1, Bb, 256, 256, 352, 0, 0, 0, fx1b, 1, 1);
  n = Bb * 256; pack_a<<<EL(n)>>>(TX1, Bb, 256, 256, TX1h, n);
  launch_gemm(stream, TX1h, fx2t, tvec, Bb, 64, 64, 256, 0, 0, 0, fx2b, 1, 1);

  // ---- head ----
  n = Bb * 128; concat_pack<<<EL(n)>>>(gvec, tvec, c16, n);
  launch_gemm(stream, c16, f12t, D1, Bb, 128, 128, 128, 0, 0, 0, f12b, 1, 1);
  n = Bb * 128; pack_a<<<EL(n)>>>(D1, Bb, 128, 128, D1h, n);
  launch_gemm(stream, D1h, f22t, D2, Bb, 64, 64, 128, 0, 0, 0, f22b, 1, 1);
  final_out<<<EL(Bb)>>>(D2, ow, ob, (float*)d_out, Bb);
#undef EL
}
